// Attention_11450382812007
// MI455X (gfx1250) — matrix operand order repaired and checked by reading, not compiled, not run on hardware
//
#include <hip/hip_runtime.h>
#include <math.h>

typedef _Float16 fp16_t;
typedef __attribute__((ext_vector_type(16))) _Float16 v16h;
typedef __attribute__((ext_vector_type(8)))  _Float16 v8h;
typedef __attribute__((ext_vector_type(8)))  float    v8f;
typedef __attribute__((ext_vector_type(4)))  int      v4i;

#define EMBED 512
#define BATCH 8
#define SEQ   2048
#define NROW  (BATCH * SEQ)

#if defined(__has_builtin)
#if __has_builtin(__builtin_amdgcn_global_load_async_to_lds_b128) && \
    __has_builtin(__builtin_amdgcn_s_wait_asynccnt)
#define USE_ASYNC_LDS 1
#endif
#endif

static __device__ __forceinline__ v16h concat8(v8h lo, v8h hi) {
    return __builtin_shufflevector(lo, hi, 0,1,2,3,4,5,6,7,8,9,10,11,12,13,14,15);
}

static __device__ __forceinline__ v8f wmma_f16(v16h a, v16h b, v8f c) {
    // D = A(16x32 f16) * B(32x16 f16) + C(16x16 f32)
    return __builtin_amdgcn_wmma_f32_16x16x32_f16(
        /*neg_a=*/false, a, /*neg_b=*/false, b,
        /*c_mod=*/(short)0, c, /*reuse_a=*/false, /*reuse_b=*/false);
}

// ---------------------------------------------------------------------------
// Kernel 1: projection  out = (x @ W^T + bias) * out_scale, fp32 -> f16.
// MODE 0/1: store row-major (B*S, E).  MODE 2: store transposed per batch (B, E, S).
// One wave computes one 16x16 output tile via 16x16x32 f16 WMMA.
// ---------------------------------------------------------------------------
template <int MODE>
__global__ __launch_bounds__(32)
void qkv_proj_kernel(const float* __restrict__ x,
                     const float* __restrict__ W,
                     const float* __restrict__ bias,
                     fp16_t* __restrict__ out,
                     float out_scale)
{
    const int lane = threadIdx.x & 31;
    const int lrow = lane & 15;          // A: row M;  B: column N
    const int g    = lane >> 4;          // A/B: K-group

    const int m0 = blockIdx.x * 16;      // row tile (over B*S)
    const int n0 = blockIdx.y * 16;      // col tile (over E)

    const float* xrow = x + (size_t)(m0 + lrow) * EMBED;
    const float* wrow = W + (size_t)(n0 + lrow) * EMBED;  // B[k][n] = W[n][k]

    v8f acc = {};
    for (int d0 = 0; d0 < EMBED; d0 += 32) {
        v16h a, b;
        // A layout: lane holds row lrow, K = d0+16g .. +15 (contiguous)
        const float* pa0 = xrow + d0 + 16 * g;
        const float* pa1 = xrow + d0 + 16 * g + 8;
        #pragma unroll
        for (int i = 0; i < 8; ++i) {
            a[i]     = (fp16_t)pa0[i];
            a[8 + i] = (fp16_t)pa1[i];
        }
        // B layout: lane holds column lrow, K = d0+16g .. +15 (contiguous W row)
        const float* pb = wrow + d0 + 16 * g;
        #pragma unroll
        for (int i = 0; i < 16; ++i) b[i] = (fp16_t)pb[i];

        acc = wmma_f16(a, b, acc);
    }

    const float bn = bias[n0 + lrow];    // column bias (N = lane%16)

    // C/D layout: VGPR r -> row M = r + 8*(lane/16), N = lane%16
    #pragma unroll
    for (int r = 0; r < 8; ++r) {
        const int row = m0 + r + 8 * g;
        const int col = n0 + lrow;
        const float v = (acc[r] + bn) * out_scale;
        if (MODE < 2) {
            out[(size_t)row * EMBED + col] = (fp16_t)v;              // (B*S, E)
        } else {
            const int b_ = row >> 11;                                 // /SEQ
            const int s_ = row & (SEQ - 1);
            out[((size_t)b_ * EMBED + col) * SEQ + s_] = (fp16_t)v;   // (B, E, S)
        }
    }
}

// ---------------------------------------------------------------------------
// Kernel 2: flash attention.  Block = 8 waves = 32 queries (2 q-tiles).
//   wave (qt, kw):  S keys [kb+32*kw, +32) of q-tile qt, and PV output
//   columns [128*kw, +128) of q-tile qt.  Doubling queries per block halves
//   K/V L2 traffic per query vs a 16-query block.
// ---------------------------------------------------------------------------
__global__ __launch_bounds__(256)
void flash_attn_kernel(const fp16_t* __restrict__ Q,   // (B*S, E), pre-scaled by 1/sqrt(E)
                       const fp16_t* __restrict__ K,   // (B*S, E)
                       const fp16_t* __restrict__ Vt,  // (B, E, S)
                       float* __restrict__ out)        // (B, S, E) fp32
{
    __shared__ __attribute__((aligned(16))) fp16_t sQ[2][16 * EMBED];  // 32 KB
    __shared__ __attribute__((aligned(16))) fp16_t sP[2][4][16 * 32];  // 8 KB
    __shared__ float sRedMax[2][4][16];
    __shared__ float sRedSum[2][4][16];

    const int tid  = threadIdx.x;
    const int w    = tid >> 5;
    const int qt   = w >> 2;      // which 16-query tile
    const int kw   = w & 3;       // key-subset index == E-column-subset index
    const int lane = tid & 31;
    const int lrow = lane & 15;
    const int g    = lane >> 4;

    const int b_ = blockIdx.y;
    const int q0 = blockIdx.x * 32;
    const size_t rowbase = (size_t)b_ * SEQ + q0 + qt * 16;

    // ---- stage 32x512 Q tile into LDS (async global->LDS when available) ----
    {
        const fp16_t* qsrc = Q + ((size_t)b_ * SEQ + q0) * EMBED;
#ifdef USE_ASYNC_LDS
        for (int i = tid; i < (32 * EMBED) / 8; i += 256) {
            __builtin_amdgcn_global_load_async_to_lds_b128(
                (__attribute__((address_space(1))) v4i*)(qsrc + (size_t)i * 8),
                (__attribute__((address_space(3))) v4i*)(&sQ[0][0] + (size_t)i * 8),
                /*offset=*/0, /*cpol=*/0);
        }
        __builtin_amdgcn_s_wait_asynccnt(0);
#else
        const v8h* src = (const v8h*)qsrc;
        v8h* dst = (v8h*)&sQ[0][0];
        for (int i = tid; i < (32 * EMBED) / 8; i += 256) dst[i] = src[i];
#endif
    }
    __syncthreads();

    float m_run[8], l_run[8];
    #pragma unroll
    for (int r = 0; r < 8; ++r) { m_run[r] = -__builtin_inff(); l_run[r] = 0.0f; }

    v8f oacc[8];
    #pragma unroll
    for (int c = 0; c < 8; ++c) oacc[c] = (v8f){};

    const fp16_t* Kbase = K  + (size_t)b_ * SEQ * EMBED;
    const fp16_t* Vbase = Vt + (size_t)b_ * EMBED * SEQ;
    const fp16_t* sQq   = &sQ[qt][0];

    for (int kb = 0; kb < SEQ; kb += 128) {
        const int key0 = kb + 32 * kw;

        const fp16_t* kcol0 = Kbase + (size_t)(key0 + lrow) * EMBED;        // B: N = key
        const fp16_t* kcol1 = Kbase + (size_t)(key0 + 16 + lrow) * EMBED;

        // prefetch next iteration's K rows into cache while we compute
        if (kb + 128 < SEQ) {
            __builtin_prefetch(kcol0 + (size_t)128 * EMBED, 0, 1);
            __builtin_prefetch(kcol1 + (size_t)128 * EMBED, 0, 1);
        }

        // ---- S tile: Q (16x512) @ K^T -> 16 x 32 for this wave's keys ----
        v8f sa0 = {}, sa1 = {};
        for (int d0 = 0; d0 < EMBED; d0 += 32) {
            v16h a  = concat8(*(const v8h*)(sQq + lrow * EMBED + d0 + 16 * g),
                              *(const v8h*)(sQq + lrow * EMBED + d0 + 16 * g + 8));
            v16h b0 = concat8(*(const v8h*)(kcol0 + d0 + 16 * g),
                              *(const v8h*)(kcol0 + d0 + 16 * g + 8));
            v16h b1 = concat8(*(const v8h*)(kcol1 + d0 + 16 * g),
                              *(const v8h*)(kcol1 + d0 + 16 * g + 8));
            sa0 = wmma_f16(a, b0, sa0);
            sa1 = wmma_f16(a, b1, sa1);
        }

        // ---- per-row max across this wave's 32 keys (16-lane groups) ----
        float lmax[8];
        #pragma unroll
        for (int r = 0; r < 8; ++r) {
            float v = fmaxf(sa0[r], sa1[r]);
            v = fmaxf(v, __shfl_xor(v, 1));
            v = fmaxf(v, __shfl_xor(v, 2));
            v = fmaxf(v, __shfl_xor(v, 4));
            v = fmaxf(v, __shfl_xor(v, 8));
            lmax[r] = v;
        }
        if (lrow == 0) {
            #pragma unroll
            for (int r = 0; r < 8; ++r) sRedMax[qt][kw][r + 8 * g] = lmax[r];
        }
        __syncthreads();   // barrier 1

        // ---- online softmax: new max, alpha, P = exp(S - m_new) ----
        float m_new[8], alpha[8], lsum[8];
        #pragma unroll
        for (int r = 0; r < 8; ++r) {
            float mn = m_run[r];
            #pragma unroll
            for (int ww = 0; ww < 4; ++ww) mn = fmaxf(mn, sRedMax[qt][ww][r + 8 * g]);
            m_new[r] = mn;
            alpha[r] = __expf(m_run[r] - mn);

            const float p0 = __expf(sa0[r] - mn);
            const float p1 = __expf(sa1[r] - mn);
            const int row = r + 8 * g;
            sP[qt][kw][row * 32 + lrow]      = (fp16_t)p0;   // A-operand row-major
            sP[qt][kw][row * 32 + 16 + lrow] = (fp16_t)p1;
            float v = p0 + p1;
            v += __shfl_xor(v, 1);
            v += __shfl_xor(v, 2);
            v += __shfl_xor(v, 4);
            v += __shfl_xor(v, 8);
            lsum[r] = v;
        }
        if (lrow == 0) {
            #pragma unroll
            for (int r = 0; r < 8; ++r) sRedSum[qt][kw][r + 8 * g] = lsum[r];
        }
        __syncthreads();   // barrier 2

        // ---- running sum update + rescale O ----
        #pragma unroll
        for (int r = 0; r < 8; ++r) {
            float s = 0.0f;
            #pragma unroll
            for (int ww = 0; ww < 4; ++ww) s += sRedSum[qt][ww][r + 8 * g];
            l_run[r] = l_run[r] * alpha[r] + s;
            m_run[r] = m_new[r];
        }
        #pragma unroll
        for (int c = 0; c < 8; ++c)
            #pragma unroll
            for (int r = 0; r < 8; ++r) oacc[c][r] *= alpha[r];

        // ---- O += P (16x128) @ V (128 x this wave's 128 cols) ----
        const int e0 = 128 * kw;
        #pragma unroll
        for (int t = 0; t < 4; ++t) {
            v16h pa = concat8(*(const v8h*)(&sP[qt][t][lrow * 32 + 16 * g]),
                              *(const v8h*)(&sP[qt][t][lrow * 32 + 16 * g + 8]));
            #pragma unroll
            for (int c = 0; c < 8; ++c) {
                const fp16_t* vb =
                    Vbase + (size_t)(e0 + 16 * c + lrow) * SEQ + kb + 32 * t + 16 * g;
                v16h bfrag = concat8(*(const v8h*)vb, *(const v8h*)(vb + 8));
                oacc[c] = wmma_f16(pa, bfrag, oacc[c]);
            }
        }
    }

    // ---- normalize and store fp32 output ----
    float* orow = out + rowbase * EMBED;
    #pragma unroll
    for (int r = 0; r < 8; ++r) {
        const float inv = 1.0f / l_run[r];
        const int row = r + 8 * g;
        #pragma unroll
        for (int c = 0; c < 8; ++c)
            orow[(size_t)row * EMBED + 128 * kw + 16 * c + lrow] = oacc[c][r] * inv;
    }
}

// ---------------------------------------------------------------------------
extern "C" void kernel_launch(void* const* d_in, const int* in_sizes, int n_in,
                              void* d_out, int out_size, void* d_ws, size_t ws_size,
                              hipStream_t stream) {
    const float* x  = (const float*)d_in[0];
    const float* Wq = (const float*)d_in[1];
    const float* bq = (const float*)d_in[2];
    const float* Wk = (const float*)d_in[3];
    const float* bk = (const float*)d_in[4];
    const float* Wv = (const float*)d_in[5];
    const float* bv = (const float*)d_in[6];
    float* outp = (float*)d_out;

    // Workspace: Q, K row-major f16; V transposed per batch (B, E, S) f16.
    fp16_t* Qf = (fp16_t*)d_ws;
    fp16_t* Kf = Qf + (size_t)NROW * EMBED;
    fp16_t* Vt = Kf + (size_t)NROW * EMBED;

    const float scale = 1.0f / sqrtf((float)EMBED);  // folded into Q

    dim3 g1(NROW / 16, EMBED / 16, 1);
    qkv_proj_kernel<0><<<g1, 32, 0, stream>>>(x, Wq, bq, Qf, scale);
    qkv_proj_kernel<1><<<g1, 32, 0, stream>>>(x, Wk, bk, Kf, 1.0f);
    qkv_proj_kernel<2><<<g1, 32, 0, stream>>>(x, Wv, bv, Vt, 1.0f);

    dim3 g2(SEQ / 32, BATCH, 1);
    flash_attn_kernel<<<g2, 256, 0, stream>>>(Qf, Kf, Vt, outp);
}
